// SpecAttentionLayer_35476429865426
// MI455X (gfx1250) — matrix operand order repaired and checked by reading, not compiled, not run on hardware
//
#include <hip/hip_runtime.h>
#include <hip/hip_fp16.h>

typedef __attribute__((ext_vector_type(16))) _Float16 v16h;
typedef __attribute__((ext_vector_type(8)))  float    v8f;

#define TSEQ 64
#define FDIM 128
#define HDIM 256

// LDS strides (elements) -- padded for conflict-free fragment access
#define TI16_LD 264   // f16 ti row-major [64][264]
#define TIT_LD  66    // f16 ti transposed [256][66]
#define P16_LD  264   // f16 P = ti@M [64][264]
#define SC_LD   65    // f32 scores [64][65]
#define ATT_LD  66    // f16 attention [64][66]

// dynamic LDS layout (bytes)
#define OFF_TI32 0
#define SZ_TI32  (TSEQ * HDIM * 4)            // 65536
#define OFF_TI16 (OFF_TI32 + SZ_TI32)
#define SZ_TI16  (TSEQ * TI16_LD * 2)         // 33792
#define OFF_TIT  (OFF_TI16 + SZ_TI16)
#define SZ_TIT   (HDIM * TIT_LD * 2)          // 33792
#define OFF_P16  (OFF_TIT + SZ_TIT)           // P region reused for att after scores
#define SZ_P16   (TSEQ * P16_LD * 2)          // 33792
#define OFF_ATTC OFF_P16
#define SZ_ATT   (TSEQ * ATT_LD * 2)          // 8448
#define OFF_ATTU (OFF_ATTC + SZ_ATT)
#define OFF_SC   (OFF_P16 + SZ_P16)
#define SZ_SC    (TSEQ * SC_LD * 4)           // 16640
#define LDS_BYTES (OFF_SC + SZ_SC)            // 183552 < 320KB/WGP

// ---- WMMA fragment helpers (contiguous k = 16*(lane/16)+i, same in A and B; wave32) ----

// 16-bit A matrix 16x32: lanes 0-15 row M=lane, K = k0+{0..15};
// lanes 16-31 same rows, K = k0+{16..31}.
__device__ __forceinline__ v16h load_a16(const _Float16* p, int ld, int r0, int k0, int lane) {
  const int hh = lane >> 4;
  const _Float16* s = p + (r0 + (lane & 15)) * ld + k0 + hh * 16;
  v16h a;
#pragma unroll
  for (int i = 0; i < 16; ++i) a[i] = s[i];
  return a;
}

// 16-bit B matrix 32x16: lane = output column N (mod 16); lanes 0-15 hold K=k0+0..15,
// lanes 16-31 hold K=k0+16..31.  p is [N][K] layout (row stride ld over N).
__device__ __forceinline__ v16h load_b16(const _Float16* p, int ld, int n0, int k0, int lane) {
  const _Float16* s = p + (n0 + (lane & 15)) * ld + k0 + ((lane >> 4) << 4);
  v16h b;
#pragma unroll
  for (int i = 0; i < 16; ++i) b[i] = s[i];
  return b;
}

__device__ __forceinline__ v8f wmma16(v16h a, v16h b, v8f c) {
  return __builtin_amdgcn_wmma_f32_16x16x32_f16(false, a, false, b, (short)0, c, false, false);
}

// ---- Kernel 1: Mt[g][h] = (Qw @ Kw^T)[h][g] / sqrt(T), stored f16 ----
__global__ __launch_bounds__(HDIM) void build_mt_kernel(
    const float* __restrict__ Qw, const float* __restrict__ Kw, _Float16* __restrict__ Mt) {
  __shared__ float krow[HDIM];
  const int g = blockIdx.x, h = threadIdx.x;
  krow[h] = Kw[g * HDIM + h];
  __syncthreads();
  const float* q = Qw + (long long)h * HDIM;
  float acc = 0.f;
#pragma unroll 4
  for (int d = 0; d < HDIM; ++d) acc = fmaf(q[d], krow[d], acc);
  Mt[g * HDIM + h] = (_Float16)(acc * 0.125f);   // 1/sqrt(T)=1/8, TEMPERATURE=1
}

// ---- Kernel 2: one workgroup (8 wave32) per batch element ----
__global__ __launch_bounds__(256) void spec_attn_kernel(
    const float* __restrict__ inR, const float* __restrict__ inI,
    const float* __restrict__ pos, const _Float16* __restrict__ Mt,
    float* __restrict__ out, int N) {
  extern __shared__ char smem[];
  float*    ti32 = (float*)(smem + OFF_TI32);
  _Float16* ti16 = (_Float16*)(smem + OFF_TI16);
  _Float16* tit  = (_Float16*)(smem + OFF_TIT);
  _Float16* p16  = (_Float16*)(smem + OFF_P16);
  _Float16* attc = (_Float16*)(smem + OFF_ATTC);
  _Float16* attu = (_Float16*)(smem + OFF_ATTU);
  float*    sc   = (float*)(smem + OFF_SC);

  const int tid  = threadIdx.x;
  const int lane = tid & 31;
  const int w    = tid >> 5;
  const long long n = blockIdx.x;

  // Phase 1: ti = concat(real, imag) + pos_emb ; stage fp32 + f16 + f16-transposed
  {
    const float* baseR = inR + n * (TSEQ * FDIM);
    const float* baseI = inI + n * (TSEQ * FDIM);
#pragma unroll 4
    for (int i = 0; i < (TSEQ * HDIM) / 256; ++i) {
      const int idx = tid + i * 256;
      const int t = idx >> 8, c = idx & 255;
      const float v = ((c < FDIM) ? baseR[t * FDIM + c] : baseI[t * FDIM + c - FDIM])
                      + pos[t * HDIM + c];
      ti32[t * HDIM + c]  = v;
      const _Float16 hv = (_Float16)v;
      ti16[t * TI16_LD + c] = hv;
      tit[c * TIT_LD + t]   = hv;   // stride 66 -> conflict-free transpose store
    }
  }
  __syncthreads();

  // Phase 2: P = ti @ M   (each wave owns column tiles {2w,2w+1} x all 4 row tiles;
  //                        each Mt B-fragment loaded exactly once per block)
  {
    v8f acc[2][4];
#pragma unroll
    for (int c = 0; c < 2; ++c)
#pragma unroll
      for (int r = 0; r < 4; ++r) acc[c][r] = (v8f)0.f;

    for (int kb = 0; kb < 8; ++kb) {
      const int k0 = kb * 32;
      v16h a[4];
#pragma unroll
      for (int r = 0; r < 4; ++r) a[r] = load_a16(ti16, TI16_LD, r * 16, k0, lane);
#pragma unroll
      for (int c = 0; c < 2; ++c) {
        const int ct = w * 2 + c;
        const v16h b = load_b16(Mt, HDIM, ct * 16, k0, lane);   // L2-resident global
#pragma unroll
        for (int r = 0; r < 4; ++r) acc[c][r] = wmma16(a[r], b, acc[c][r]);
      }
    }
    const int hh = (lane >> 4) << 3, cl = lane & 15;
#pragma unroll
    for (int c = 0; c < 2; ++c)
#pragma unroll
      for (int r = 0; r < 4; ++r)
#pragma unroll
        for (int g = 0; g < 8; ++g)
          p16[(r * 16 + hh + g) * P16_LD + (w * 2 + c) * 16 + cl] = (_Float16)acc[c][r][g];
  }
  __syncthreads();

  // Phase 3: scores = P @ ti^T  (16 tiles, 2 per wave)
  {
    const int rt = w >> 1;
    v8f acc[2];
#pragma unroll
    for (int c = 0; c < 2; ++c) acc[c] = (v8f)0.f;
    for (int kb = 0; kb < 8; ++kb) {
      const int k0 = kb * 32;
      const v16h a = load_a16(p16, P16_LD, rt * 16, k0, lane);
#pragma unroll
      for (int c = 0; c < 2; ++c) {
        const int ct = (w & 1) * 2 + c;
        const v16h b = load_b16(ti16, TI16_LD, ct * 16, k0, lane); // B[k][s] = ti[s][k]
        acc[c] = wmma16(a, b, acc[c]);
      }
    }
    const int hh = (lane >> 4) << 3, cl = lane & 15;
#pragma unroll
    for (int c = 0; c < 2; ++c)
#pragma unroll
      for (int g = 0; g < 8; ++g)
        sc[(rt * 16 + hh + g) * SC_LD + ((w & 1) * 2 + c) * 16 + cl] = acc[c][g];
  }
  __syncthreads();

  // Phase 4: dual softmax (rows); att buffers overwrite the P region
  if (tid < TSEQ) {
    const float* row = sc + tid * SC_LD;
    float mx = -3.4e38f, mn = 3.4e38f;
    for (int s = 0; s < TSEQ; ++s) { const float v = row[s]; mx = fmaxf(mx, v); mn = fminf(mn, v); }
    float sumc = 0.f, sumu = 0.f;
    for (int s = 0; s < TSEQ; ++s) {
      const float v = row[s];
      const float ec = __expf(v - mx);   // softmax(+scores)
      const float eu = __expf(mn - v);   // softmax(-scores): max(-v) = -mn
      sumc += ec; sumu += eu;
      attc[tid * ATT_LD + s] = (_Float16)ec;
      attu[tid * ATT_LD + s] = (_Float16)eu;
    }
    const float rc = 1.f / sumc, ru = 1.f / sumu;
    for (int s = 0; s < TSEQ; ++s) {
      attc[tid * ATT_LD + s] = (_Float16)((float)attc[tid * ATT_LD + s] * rc);
      attu[tid * ATT_LD + s] = (_Float16)((float)attu[tid * ATT_LD + s] * ru);
    }
  }
  __syncthreads();

  // Phase 5: cau_out = att_c @ ti + ti ; spu_out = att_u @ ti
  {
    const long long spuOff = (long long)N * TSEQ * HDIM;
    const int hh = (lane >> 4) << 3, cl = lane & 15;
#pragma unroll
    for (int which = 0; which < 2; ++which) {
      const _Float16* att = which ? attu : attc;
      float* obase = out + (which ? spuOff : 0) + n * (TSEQ * HDIM);
      v8f acc[2][4];
#pragma unroll
      for (int c = 0; c < 2; ++c)
#pragma unroll
        for (int r = 0; r < 4; ++r) acc[c][r] = (v8f)0.f;
#pragma unroll
      for (int kb = 0; kb < 2; ++kb) {
        const int k0 = kb * 32;
        v16h a[4];
#pragma unroll
        for (int r = 0; r < 4; ++r) a[r] = load_a16(att, ATT_LD, r * 16, k0, lane);
#pragma unroll
        for (int c = 0; c < 2; ++c) {
          const v16h b = load_b16(tit, TIT_LD, (w * 2 + c) * 16, k0, lane); // B[s][col]=tit[col][s]
#pragma unroll
          for (int r = 0; r < 4; ++r) acc[c][r] = wmma16(a[r], b, acc[c][r]);
        }
      }
#pragma unroll
      for (int c = 0; c < 2; ++c)
#pragma unroll
        for (int r = 0; r < 4; ++r)
#pragma unroll
          for (int g = 0; g < 8; ++g) {
            const int rr = r * 16 + hh + g;
            const int cc = (w * 2 + c) * 16 + cl;
            float v = acc[c][r][g];
            if (which == 0) v += ti32[rr * HDIM + cc];   // exact fp32 residual
            obase[rr * HDIM + cc] = v;
          }
    }
  }
}

extern "C" void kernel_launch(void* const* d_in, const int* in_sizes, int n_in,
                              void* d_out, int out_size, void* d_ws, size_t ws_size,
                              hipStream_t stream) {
  const float* inR = (const float*)d_in[0];
  const float* inI = (const float*)d_in[1];
  const float* Qw  = (const float*)d_in[2];
  const float* Kw  = (const float*)d_in[3];
  const float* pos = (const float*)d_in[4];
  const int N = in_sizes[0] / (TSEQ * FDIM);   // 4096

  _Float16* Mt = (_Float16*)d_ws;              // 256x256 f16 = 128 KB

  build_mt_kernel<<<HDIM, HDIM, 0, stream>>>(Qw, Kw, Mt);
  spec_attn_kernel<<<N, 256, LDS_BYTES, stream>>>(inR, inI, pos, Mt, (float*)d_out, N);
}
